// CoxNAM_17772574670892
// MI455X (gfx1250) — matrix operand order repaired and checked by reading, not compiled, not run on hardware
//
#include <hip/hip_runtime.h>

typedef __attribute__((ext_vector_type(16))) _Float16 v16h;
typedef __attribute__((ext_vector_type(8)))  _Float16 v8h;
typedef __attribute__((ext_vector_type(8)))  float    v8f;

#define NFEAT   256
#define H1DIM   256
#define H2DIM   128
#define BTOTAL  4096
#define KPAD    264      // 256 + 8 halfs -> 528B row stride: 16B aligned, conflict-free
#define MBLOCK  512      // rows per block

__global__ void zero_out_kernel(float* __restrict__ out, int n) {
    int i = blockIdx.x * blockDim.x + threadIdx.x;
    if (i < n) out[i] = 0.0f;
}

__global__ __launch_bounds__(256)
void nam_wmma_kernel(const float* __restrict__ x,
                     const float* __restrict__ W1,
                     const float* __restrict__ b1,
                     const float* __restrict__ W2,
                     const float* __restrict__ b2,
                     const float* __restrict__ W3,
                     const float* __restrict__ b3,
                     float* __restrict__ out)
{
    __shared__ _Float16 sW2[H2DIM * KPAD];       // [n][k], f16, padded
    __shared__ float    sW1[H1DIM];
    __shared__ float    sb1[H1DIM];
    __shared__ float    sb2[H2DIM];
    __shared__ float    sW3[H2DIM];

    const int f   = blockIdx.y;       // feature 0..255
    const int bm  = blockIdx.x;       // row-slab 0..7
    const int tid = threadIdx.x;      // 0..255 (8 waves)

    // ---- cooperative stage: per-feature params into LDS ----
    sW1[tid] = W1[f * H1DIM + tid];   // W1 is (F,1,H1)
    sb1[tid] = b1[f * H1DIM + tid];
    if (tid < H2DIM) {
        sb2[tid] = b2[f * H2DIM + tid];
        sW3[tid] = W3[f * H2DIM + tid];   // W3 is (F,H2,1)
    }
    const float b3v = b3[f];

    // W2[f] : (H1=256, H2=128) f32, row-major -> LDS as f16 [n][k]
    const float* __restrict__ W2f = W2 + (size_t)f * (H1DIM * H2DIM);
    #pragma clang loop unroll(disable)
    for (int task = tid; task < (H1DIM / 4) * H2DIM; task += 256) {
        const int n  = task & (H2DIM - 1);     // consecutive tid -> consecutive n (coalesced)
        const int kg = task >> 7;              // group of 4 k values
        const float* __restrict__ src = W2f + (4 * kg) * H2DIM + n;
        _Float16* __restrict__ dst = &sW2[n * KPAD + 4 * kg];  // 8B-aligned
        dst[0] = (_Float16)src[0];
        dst[1] = (_Float16)src[H2DIM];
        dst[2] = (_Float16)src[2 * H2DIM];
        dst[3] = (_Float16)src[3 * H2DIM];
    }
    __syncthreads();

    const int lane  = tid & 31;
    const int wave  = tid >> 5;       // 0..7
    const int nrow  = lane & 15;      // row-in-tile for A; N-in-tile for B/C/D
    const int khalf = lane >> 4;      // 0/1: K offset khalf*8 within each 16-K block (A and B)

    #pragma clang loop unroll(disable)
    for (int it = 0; it < MBLOCK / 128; ++it) {
        const int m0 = bm * MBLOCK + it * 128 + wave * 16;   // 16-row tile base
        const float xv = x[(size_t)(m0 + nrow) * NFEAT + f];

        v8f acc[8] = {};   // 8 N-tiles of 16x16 f32

        // Keep the K loop ROLLED: steady-state live set = 64 acc + 8 A + transient B
        // fragments (~140 VGPRs). Full unroll previously exploded past the 256-VGPR
        // window (s_set_vgpr_msb) and spilled B fragments to scratch.
        #pragma clang loop unroll(disable)
        for (int ks = 0; ks < 8; ++ks) {
            const int k0 = ks * 32;
            const int kb = k0 + khalf * 8;

            // Build A fragment (16x32 f16) on the fly: h1 = relu(x*W1 + b1).
            // ISA A layout: elem e<8 -> K = khalf*8+e ; e>=8 -> K = 16+khalf*8+(e-8)
            v16h a;
            #pragma unroll
            for (int e = 0; e < 8; ++e) {
                a[e]     = (_Float16)fmaxf(fmaf(xv, sW1[kb + e],      sb1[kb + e]),      0.0f);
                a[e + 8] = (_Float16)fmaxf(fmaf(xv, sW1[kb + 16 + e], sb1[kb + 16 + e]), 0.0f);
            }

            #pragma unroll
            for (int t = 0; t < 8; ++t) {
                // B fragment (32x16 f16): same K order as A.
                // lane l: n = l&15, e<8 -> K = (l>>4)*8 + e ; e>=8 -> K = 16 + (l>>4)*8 + (e-8)
                const _Float16* __restrict__ bp =
                    &sW2[(t * 16 + nrow) * KPAD + k0 + khalf * 8];
                const v8h blo = *(const v8h*)(bp);       // 16B ds_load_b128
                const v8h bhi = *(const v8h*)(bp + 16);  // 16B ds_load_b128
                v16h bfrag;
                #pragma unroll
                for (int e = 0; e < 8; ++e) { bfrag[e] = blo[e]; bfrag[e + 8] = bhi[e]; }

                acc[t] = __builtin_amdgcn_wmma_f32_16x16x32_f16(
                    /*neg_a=*/false, a, /*neg_b=*/false, bfrag,
                    /*c_mod=*/(short)0, acc[t],
                    /*reuse_a=*/false, /*reuse_b=*/false);
            }
        }

        // ---- fused epilogue: h2 = relu(acc + b2); contrib = h2 . W3 + b3 ----
        // C/D layout: VGPR r -> M = r + 8*khalf ; lane -> N = t*16 + (lane&15)
        float c[8];
        #pragma unroll
        for (int r = 0; r < 8; ++r) c[r] = 0.0f;
        #pragma unroll
        for (int t = 0; t < 8; ++t) {
            const float b2v = sb2[t * 16 + nrow];
            const float w3v = sW3[t * 16 + nrow];
            #pragma unroll
            for (int r = 0; r < 8; ++r)
                c[r] = fmaf(fmaxf(acc[t][r] + b2v, 0.0f), w3v, c[r]);
        }

        // Reduce over N within each 16-lane half (xor masks stay inside the half),
        // then one atomic per row per feature.
        #pragma unroll
        for (int r = 0; r < 8; ++r) {
            float v = c[r];
            v += __shfl_xor(v, 1, 32);
            v += __shfl_xor(v, 2, 32);
            v += __shfl_xor(v, 4, 32);
            v += __shfl_xor(v, 8, 32);
            if (nrow == r) {   // lanes 0..7 -> M=r ; lanes 16..23 -> M=r+8
                atomicAdd(&out[m0 + r + 8 * khalf], v + b3v);
            }
        }
    }
}

extern "C" void kernel_launch(void* const* d_in, const int* in_sizes, int n_in,
                              void* d_out, int out_size, void* d_ws, size_t ws_size,
                              hipStream_t stream) {
    (void)in_sizes; (void)n_in; (void)d_ws; (void)ws_size;
    const float* x  = (const float*)d_in[0];
    const float* W1 = (const float*)d_in[1];
    const float* b1 = (const float*)d_in[2];
    const float* W2 = (const float*)d_in[3];
    const float* b2 = (const float*)d_in[4];
    const float* W3 = (const float*)d_in[5];
    const float* b3 = (const float*)d_in[6];
    float* out = (float*)d_out;

    zero_out_kernel<<<(BTOTAL + 255) / 256, 256, 0, stream>>>(out, out_size);

    dim3 grid(BTOTAL / MBLOCK, NFEAT);   // (8, 256)
    nam_wmma_kernel<<<grid, 256, 0, stream>>>(x, W1, b1, W2, b2, W3, b3, out);
}
